// SpatialMHA_44109314130347
// MI455X (gfx1250) — matrix operand order repaired and checked by reading, not compiled, not run on hardware
//
#include <hip/hip_runtime.h>
#include <hip/hip_bf16.h>
#include <cstdint>
#include <cstddef>

// Problem constants (match reference)
#define B_ 128
#define E_ 512
#define C_ 128
#define H_ 8
#define D_ 64

typedef __attribute__((ext_vector_type(16))) __bf16 v16bf;
typedef __attribute__((ext_vector_type(8)))  float  v8f;

__device__ inline unsigned short f2bf(float f) {
  unsigned int u = __float_as_uint(f);
  u += 0x7FFFu + ((u >> 16) & 1u);          // round-to-nearest-even
  return (unsigned short)(u >> 16);
}

__device__ inline float selu_f(float x) {
  const float l  = 1.0507009873554805f;     // lambda
  const float la = 1.7580993408473766f;     // lambda*alpha
  return x > 0.f ? l * x : la * (__expf(x) - 1.f);
}

// ---------------------------------------------------------------------------
// CDNA5 async memory->LDS copy (16 bytes per lane), GVS addressing; ASYNCcnt.
// Generic __shared__ pointer low 32 bits == LDS byte offset (aperture rule).
// ---------------------------------------------------------------------------
__device__ __forceinline__ void async_cp16(void* lds_dst, const void* gbase,
                                           uint32_t byte_off) {
  uint32_t ldsoff = (uint32_t)(uintptr_t)lds_dst;
  asm volatile("global_load_async_to_lds_b128 %0, %1, %2"
               :: "v"(ldsoff), "v"(byte_off), "s"(gbase)
               : "memory");
}
__device__ __forceinline__ void wait_async0() {
  asm volatile("s_wait_asynccnt 0" ::: "memory");
}

// CDNA5 WMMA bf16 fragment gathers (ISA 7.12.2 layouts, wave32).
// A (16x32, MxK): lane L: m = L&15; halves of the row at k-offsets 8*h2, 16+8*h2.
// B (32x16, KxN): lane L: n = L&15; halves of the column at k-offsets 8*h2, 16+8*h2.
union FragU { v16bf v; uint4 q[2]; };

__device__ inline v16bf frag_a(const unsigned short* rowbase, int h2) {
  FragU f;
  f.q[0] = *reinterpret_cast<const uint4*>(rowbase + 8 * h2);
  f.q[1] = *reinterpret_cast<const uint4*>(rowbase + 16 + 8 * h2);
  return f.v;
}
__device__ inline v16bf frag_b(const unsigned short* rowbase, int h2) {
  FragU f;
  f.q[0] = *reinterpret_cast<const uint4*>(rowbase + 8 * h2);
  f.q[1] = *reinterpret_cast<const uint4*>(rowbase + 16 + 8 * h2);
  return f.v;
}

// ---------------------------------------------------------------------------
// f32 -> bf16 (weights, done once)
// ---------------------------------------------------------------------------
__global__ __launch_bounds__(256) void cvt_bf16(const float* __restrict__ src,
                                                unsigned short* __restrict__ dst,
                                                int n) {
  int i = blockIdx.x * 256 + threadIdx.x;
  if (i < n) dst[i] = f2bf(src[i]);
}

// ---------------------------------------------------------------------------
// x [B,E,C] f32 -> xT [B,C,E] bf16 (transpose via LDS tile, done once)
// ---------------------------------------------------------------------------
__global__ __launch_bounds__(256) void cvt_x_T(const float* __restrict__ x,
                                               unsigned short* __restrict__ xT) {
  __shared__ unsigned short tile[128 * 32];   // [c][e]
  const int b = blockIdx.y, e0 = blockIdx.x * 32, t = threadIdx.x;
#pragma unroll
  for (int s = 0; s < 16; ++s) {              // read 32e x 128c coalesced
    int id = t + s * 256, e = id >> 7, c = id & 127;
    tile[c * 32 + e] = f2bf(x[((size_t)b * E_ + e0 + e) * C_ + c]);
  }
  __syncthreads();
#pragma unroll
  for (int s = 0; s < 16; ++s) {              // write rows of xT coalesced
    int id = t + s * 256, c = id >> 5, k = id & 31;
    xT[((size_t)b * C_ + c) * E_ + e0 + k] = tile[c * 32 + k];
  }
}

// ---------------------------------------------------------------------------
// Fused bias table biasM[h][i][j] (attn_bias + antisym dir bias, kNN -> -inf)
// ---------------------------------------------------------------------------
__global__ __launch_bounds__(128) void bias_kernel(
    const float* __restrict__ dir_vec, const float* __restrict__ attn_bias,
    const float* __restrict__ disp, const unsigned char* __restrict__ knn,
    float* __restrict__ biasM) {
  const int i = blockIdx.x, h = blockIdx.y, j = threadIdx.x;
  __shared__ float red[128];

  float dv = dir_vec[h * C_ + j];
  red[j] = dv * dv;
  __syncthreads();
  for (int s = 64; s > 0; s >>= 1) { if (j < s) red[j] += red[j + s]; __syncthreads(); }
  const float nrm = fmaxf(sqrtf(red[0]), 1e-12f);
  __syncthreads();

  const float dvn_j = dir_vec[h * C_ + j] / nrm;
  const float dvn_i = dir_vec[h * C_ + i] / nrm;
  const float third = 1.f / 3.f;
  const float dm_ij = (disp[(i * C_ + j) * 3 + 0] + disp[(i * C_ + j) * 3 + 1] +
                       disp[(i * C_ + j) * 3 + 2]) * third;
  const float dm_ji = (disp[(j * C_ + i) * 3 + 0] + disp[(j * C_ + i) * 3 + 1] +
                       disp[(j * C_ + i) * 3 + 2]) * third;
  const float anti = 0.5f * (dm_ij * dvn_j - dm_ji * dvn_i);

  red[j] = anti;
  __syncthreads();
  for (int s = 64; s > 0; s >>= 1) { if (j < s) red[j] += red[j + s]; __syncthreads(); }
  const float mean = red[0] * (1.f / (float)C_);

  float val = attn_bias[i * C_ + j] + anti - mean;
  if (!knn[i * C_ + j]) val = -__builtin_inff();
  biasM[((size_t)h * C_ + i) * C_ + j] = val;
}

// ---------------------------------------------------------------------------
// Projection GEMM: Y[b] = selu(Wbf (512x512 bf16) * XT[b]^T), XT is [B,C,E] bf16.
// 128x128 block tile, K stepped by 32, DOUBLE-BUFFERED async-to-LDS staging:
// copies for step kt+1 are issued before computing step kt, s_wait_asynccnt 0
// only at the end of the step -> DMA overlaps the 8 WMMAs per wave.
// OUTMODE 0: bf16 transposed [B,C,E] (q/k). 1: bf16 natural [B,E,C] (v).
// OUTMODE 2: f32 natural [B,E,C] (final output).
// ---------------------------------------------------------------------------
template <int OUTMODE>
__global__ __launch_bounds__(256) void gemm_proj(
    const unsigned short* __restrict__ Wbf, const unsigned short* __restrict__ XT,
    void* __restrict__ Yv) {
  // [0..4095]=A buf0, [4096..8191]=A buf1, [8192..12287]=B buf0, [12288..]=B buf1
  alignas(32) __shared__ unsigned short lds[16384];  // 32 KB; reused as out tile
  const int t = threadIdx.x, b = blockIdx.y, m0 = blockIdx.x * 128;
  const int w = t >> 5, L = t & 31, h2 = L >> 4, ln = L & 15;

  // per-thread staging chunks: tiles are 128 rows x 32 k = 512 x 16B chunks
  const int am0 = t >> 2, ach = t & 3;          // chunk t
  const int am1 = am0 + 64;                     // chunk t+256
  const int offc0 = am0 * 32 + ach * 8;         // LDS element offset (both tiles)
  const int offc1 = am1 * 32 + ach * 8;
  const uint32_t ga0 = (uint32_t)(((m0 + am0) * E_ + ach * 8) * 2);   // W bytes
  const uint32_t ga1 = (uint32_t)(((m0 + am1) * E_ + ach * 8) * 2);
  const uint32_t gb0 = (uint32_t)((((size_t)b * C_ + am0) * E_ + ach * 8) * 2);
  const uint32_t gb1 = (uint32_t)((((size_t)b * C_ + am1) * E_ + ach * 8) * 2);

  auto stage = [&](int bufi, int k0) {
    unsigned short* A  = lds + bufi * 4096;
    unsigned short* Bt = lds + 8192 + bufi * 4096;
    const uint32_t kb = (uint32_t)(k0 * 2);
    async_cp16(A + offc0,  Wbf, ga0 + kb);
    async_cp16(A + offc1,  Wbf, ga1 + kb);
    async_cp16(Bt + offc0, XT,  gb0 + kb);
    async_cp16(Bt + offc1, XT,  gb1 + kb);
  };

  v8f acc[8] = {};
  stage(0, 0);
  wait_async0();
  __syncthreads();

  for (int kt = 0; kt < E_ / 32; ++kt) {
    const int cur = kt & 1;
    if (kt + 1 < E_ / 32) stage(cur ^ 1, (kt + 1) * 32);  // overlap with compute

    const unsigned short* A  = lds + cur * 4096;
    const unsigned short* Bt = lds + 8192 + cur * 4096;
    v16bf af = frag_a(&A[(w * 16 + ln) * 32], h2);
#pragma unroll
    for (int nt = 0; nt < 8; ++nt) {
      v16bf bf = frag_b(&Bt[(nt * 16 + ln) * 32], h2);
      acc[nt] = __builtin_amdgcn_wmma_f32_16x16x32_bf16(
          false, af, false, bf, (short)0, acc[nt], false, false);
    }
    wait_async0();      // next-step copies landed (hidden behind the WMMAs)
    __syncthreads();
  }

  if constexpr (OUTMODE == 2) {
    float* Y = (float*)Yv;
#pragma unroll
    for (int nt = 0; nt < 8; ++nt)
#pragma unroll
      for (int r = 0; r < 8; ++r) {
        const int o = m0 + w * 16 + 8 * h2 + r;
        const int c = nt * 16 + ln;
        Y[(size_t)b * E_ * C_ + (size_t)o * C_ + c] = selu_f(acc[nt][r]);
      }
  } else {
    // restage 128x128 bf16 out tile in LDS, then 16B coalesced global stores
    unsigned short* sT = lds;
#pragma unroll
    for (int nt = 0; nt < 8; ++nt)
#pragma unroll
      for (int r = 0; r < 8; ++r) {
        const int ol = w * 16 + 8 * h2 + r;     // local o
        const int c  = nt * 16 + ln;
        const unsigned short bv = f2bf(selu_f(acc[nt][r]));
        if constexpr (OUTMODE == 0) sT[c * 128 + ol] = bv;   // [c][o]
        else                        sT[ol * 128 + c] = bv;   // [o][c]
      }
    __syncthreads();
    unsigned short* Y = (unsigned short*)Yv;
#pragma unroll
    for (int s = 0; s < 8; ++s) {
      const int id = t + s * 256, row = id >> 4, ch = id & 15;
      size_t dst;
      if constexpr (OUTMODE == 0)      // Y[b][c=row][m0+..]  (transposed)
        dst = ((size_t)b * C_ + row) * E_ + m0 + ch * 8;
      else                             // Y[b][m0+row][..]    (natural)
        dst = ((size_t)b * E_ + m0 + row) * C_ + ch * 8;
      *reinterpret_cast<uint4*>(&Y[dst]) =
          *reinterpret_cast<const uint4*>(&sT[row * 128 + ch * 8]);
    }
  }
}

// ---------------------------------------------------------------------------
// Per-(b,h) attention. S = q^T k / 8 + biasM -> register softmax -> Out = V P^T.
// q/k arrive pre-transposed ([B,C,E]) and v natural, so ALL staging is pure
// global_load_async_to_lds_b128. Output stored transposed abufT[b][c][e].
// ---------------------------------------------------------------------------
__global__ __launch_bounds__(256) void attn_kernel(
    const unsigned short* __restrict__ qT, const unsigned short* __restrict__ kT,
    const unsigned short* __restrict__ vbuf, const float* __restrict__ biasM,
    unsigned short* __restrict__ abufT) {
  alignas(32) __shared__ unsigned short sAB[2 * 128 * 64]; // qt[i][d] | kt[j][d]; reused as p[i][j]
  alignas(32) __shared__ unsigned short sV[64 * 128];      // v[d][j]; reused as out[i][d]
  const int t = threadIdx.x;
  const int b = blockIdx.x, h = blockIdx.y;
  const size_t base = ((size_t)(b * H_ + h)) * D_ * C_;    // v slice (natural)
  unsigned short* sQT = sAB;
  unsigned short* sKT = sAB + 128 * 64;

  // all three operands staged via async copies (12 x 16B per thread)
#pragma unroll
  for (int s = 0; s < 4; ++s) {
    const int id = t + s * 256;                 // 1024 chunks per operand
    async_cp16(&sV[id * 8], vbuf, (uint32_t)((base + (size_t)id * 8) * 2));
    const int i = id >> 3, ch = id & 7;         // qT/kT rows: 64 d's = 8 chunks
    const uint32_t qko =
        (uint32_t)(((((size_t)b * C_ + i) * E_) + h * D_ + ch * 8) * 2);
    async_cp16(&sQT[id * 8], qT, qko);
    async_cp16(&sKT[id * 8], kT, qko);
  }
  wait_async0();
  __syncthreads();

  const int w = t >> 5, L = t & 31, h2 = L >> 4, ln = L & 15;

  // ---- phase 1: scores. Wave w owns rows i in [16w,16w+16), all 8 j-subtiles.
  v8f acc[8] = {};
  const int m0 = w * 16;
#pragma unroll
  for (int kk = 0; kk < 2; ++kk) {
    const int d0 = kk * 32;
    v16bf af = frag_a(&sQT[(m0 + ln) * 64 + d0], h2);
#pragma unroll
    for (int nt = 0; nt < 8; ++nt) {
      v16bf bf = frag_b(&sKT[(nt * 16 + ln) * 64 + d0], h2);
      acc[nt] = __builtin_amdgcn_wmma_f32_16x16x32_bf16(
          false, af, false, bf, (short)0, acc[nt], false, false);
    }
  }
  __syncthreads();  // all q/k LDS reads done before p overwrites sAB

  // ---- softmax in registers; row i = m0 + 8*h2 + r lives on lanes {h2*16+0..15}
  unsigned short* sP = sAB;  // p[i][j] bf16
#pragma unroll
  for (int r = 0; r < 8; ++r) {
    const int i = m0 + 8 * h2 + r;
    const float* gb = biasM + ((size_t)h * C_ + i) * C_;
    float v[8];
    float mx = -__builtin_inff();
#pragma unroll
    for (int nt = 0; nt < 8; ++nt) {
      v[nt] = acc[nt][r] * 0.125f + gb[nt * 16 + ln];  // scale=D^-0.5, masked -> -inf
      mx = fmaxf(mx, v[nt]);
    }
#pragma unroll
    for (int mk = 1; mk <= 8; mk <<= 1) mx = fmaxf(mx, __shfl_xor(mx, mk, 32));
    float sum = 0.f;
#pragma unroll
    for (int nt = 0; nt < 8; ++nt) { v[nt] = __expf(v[nt] - mx); sum += v[nt]; }
#pragma unroll
    for (int mk = 1; mk <= 8; mk <<= 1) sum += __shfl_xor(sum, mk, 32);
    const float inv = 1.f / sum;
#pragma unroll
    for (int nt = 0; nt < 8; ++nt) sP[i * 128 + nt * 16 + ln] = f2bf(v[nt] * inv);
  }
  __syncthreads();

  // ---- phase 2: Out[d][i] = sum_j v[d][j] * p[i][j]  (A = V, B = P^T)
  v8f acc2[4] = {};
  const int md = (w & 3) * 16;
  const int nb = (w >> 2) * 64;
#pragma unroll
  for (int kk = 0; kk < 4; ++kk) {
    const int k0 = kk * 32;
    v16bf af = frag_a(&sV[(md + ln) * 128 + k0], h2);
#pragma unroll
    for (int nt = 0; nt < 4; ++nt) {
      v16bf bf = frag_b(&sP[(nb + nt * 16 + ln) * 128 + k0], h2);
      acc2[nt] = __builtin_amdgcn_wmma_f32_16x16x32_bf16(
          false, af, false, bf, (short)0, acc2[nt], false, false);
    }
  }
  __syncthreads();                 // sV frag reads done; reuse as out tile [i][d]
  unsigned short* sOut = sV;
#pragma unroll
  for (int nt = 0; nt < 4; ++nt)
#pragma unroll
    for (int r = 0; r < 8; ++r) {
      const int d = md + 8 * h2 + r;
      const int i = nb + nt * 16 + ln;
      sOut[i * 64 + d] = f2bf(acc2[nt][r]);
    }
  __syncthreads();
  // coalesced 16B stores: abufT[b][i][h*64 + 0..31]
#pragma unroll
  for (int s = 0; s < 4; ++s) {
    const int id = t + s * 256, i = id >> 2, ch = id & 3;
    *reinterpret_cast<uint4*>(&abufT[((size_t)b * C_ + i) * E_ + h * D_ + ch * 8]) =
        *reinterpret_cast<const uint4*>(&sOut[i * 64 + ch * 8]);
  }
}

// ---------------------------------------------------------------------------
extern "C" void kernel_launch(void* const* d_in, const int* in_sizes, int n_in,
                              void* d_out, int out_size, void* d_ws, size_t ws_size,
                              hipStream_t stream) {
  const float* x         = (const float*)d_in[0];
  const float* wq        = (const float*)d_in[1];
  const float* wk        = (const float*)d_in[2];
  const float* wv        = (const float*)d_in[3];
  const float* wo        = (const float*)d_in[4];
  const float* dir_vec   = (const float*)d_in[5];
  const float* attn_bias = (const float*)d_in[6];
  const float* disp      = (const float*)d_in[7];
  const unsigned char* knn = (const unsigned char*)d_in[8];
  (void)in_sizes; (void)n_in; (void)out_size; (void)ws_size;

  // Workspace carve (bytes):
  //  0..16M qT | 16..32M kT | 32..48M vbuf | 48..64M abufT | 64..80M xT
  //  80M biasM (512K) | 80.5M..82.5M weights bf16 (4 x 512K)
  char* ws = (char*)d_ws;
  unsigned short* qT    = (unsigned short*)(ws + ((size_t)0  << 20));
  unsigned short* kT    = (unsigned short*)(ws + ((size_t)16 << 20));
  unsigned short* vbuf  = (unsigned short*)(ws + ((size_t)32 << 20));
  unsigned short* abufT = (unsigned short*)(ws + ((size_t)48 << 20));
  unsigned short* xT    = (unsigned short*)(ws + ((size_t)64 << 20));
  float*          biasM = (float*)(ws + ((size_t)80 << 20));
  unsigned short* wqb   = (unsigned short*)(ws + ((size_t)80 << 20) + ((size_t)1 << 19));
  unsigned short* wkb   = wqb + (size_t)E_ * E_;
  unsigned short* wvb   = wkb + (size_t)E_ * E_;
  unsigned short* wob   = wvb + (size_t)E_ * E_;

  const int NW = E_ * E_;
  cvt_bf16<<<dim3(NW / 256), 256, 0, stream>>>(wq, wqb, NW);
  cvt_bf16<<<dim3(NW / 256), 256, 0, stream>>>(wk, wkb, NW);
  cvt_bf16<<<dim3(NW / 256), 256, 0, stream>>>(wv, wvb, NW);
  cvt_bf16<<<dim3(NW / 256), 256, 0, stream>>>(wo, wob, NW);
  cvt_x_T<<<dim3(E_ / 32, B_), 256, 0, stream>>>(x, xT);
  bias_kernel<<<dim3(C_, H_), 128, 0, stream>>>(dir_vec, attn_bias, disp, knn, biasM);

  gemm_proj<0><<<dim3(E_ / 128, B_), 256, 0, stream>>>(wqb, xT, qT);
  gemm_proj<0><<<dim3(E_ / 128, B_), 256, 0, stream>>>(wkb, xT, kT);
  gemm_proj<1><<<dim3(E_ / 128, B_), 256, 0, stream>>>(wvb, xT, vbuf);
  attn_kernel<<<dim3(B_, H_), 256, 0, stream>>>(qT, kT, vbuf, biasM, abufT);
  gemm_proj<2><<<dim3(E_ / 128, B_), 256, 0, stream>>>(wob, abufT, (float*)d_out);
}
